// GINVirtual_node_19602230739676
// MI455X (gfx1250) — matrix operand order repaired and checked by reading, not compiled, not run on hardware
//
#include <hip/hip_runtime.h>
#include <hip/hip_bf16.h>
#include <stdint.h>

#define EMB      300
#define HID      600
#define NNODES   60000
#define NEDGES   150000
#define NGRAPHS  2500
#define NLAYERS  5
#define KPAD1    320   // pad of K=300 to multiple of 32
#define KPAD2    608   // pad of K=600 to multiple of 32
#define BN_SCALE 0.99999500003749971f  // 1/sqrt(1+1e-5)

typedef __attribute__((ext_vector_type(16))) __bf16 v16bf;
typedef __attribute__((ext_vector_type(8)))  __bf16 v8bf;
typedef __attribute__((ext_vector_type(4)))  __bf16 v4bf;
typedef __attribute__((ext_vector_type(8)))  float  v8f;

// ---------------------------------------------------------------- small utils
__global__ void k_count(const int* __restrict__ batch, float* __restrict__ counts) {
  int n = blockIdx.x * blockDim.x + threadIdx.x;
  if (n < NNODES) atomicAdd(&counts[batch[n]], 1.0f);
}

__global__ void k_invcounts(const float* __restrict__ counts, float* __restrict__ inv) {
  int g = blockIdx.x * blockDim.x + threadIdx.x;
  if (g < NGRAPHS) inv[g] = 1.0f / fmaxf(counts[g], 1.0f);
}

__global__ void k_vn_init(const float* __restrict__ vn_emb, float* __restrict__ vn) {
  int idx = blockIdx.x * blockDim.x + threadIdx.x;
  if (idx < NGRAPHS * EMB) vn[idx] = vn_emb[idx % EMB];
}

// h_in = h + vn[batch], vectorized x4 (EMB=300 -> 75 float4 per row)
__global__ void k_addvn(const float* __restrict__ h, const float* __restrict__ vn,
                        const int* __restrict__ batch, float* __restrict__ h_in) {
  int idx = blockIdx.x * blockDim.x + threadIdx.x;
  if (idx < NNODES * 75) {
    int n = idx / 75, q = idx - n * 75;
    const float4 a = *(const float4*)(h + (size_t)n * EMB + q * 4);
    const float4 b = *(const float4*)(vn + (size_t)batch[n] * EMB + q * 4);
    float4 r; r.x = a.x + b.x; r.y = a.y + b.y; r.z = a.z + b.z; r.w = a.w + b.w;
    *(float4*)(h_in + (size_t)n * EMB + q * 4) = r;
  }
}

__global__ void k_pool(const float* __restrict__ src, const int* __restrict__ batch,
                       float* __restrict__ dst) {
  int idx = blockIdx.x * blockDim.x + threadIdx.x;
  if (idx < NNODES * 75) {
    int n = idx / 75, q = idx - n * 75;
    const float4 a = *(const float4*)(src + (size_t)n * EMB + q * 4);
    float* d = dst + (size_t)batch[n] * EMB + q * 4;
    atomicAdd(d + 0, a.x); atomicAdd(d + 1, a.y);
    atomicAdd(d + 2, a.z); atomicAdd(d + 3, a.w);
  }
}

__global__ void k_scale_rows(float* __restrict__ buf, const float* __restrict__ inv) {
  int idx = blockIdx.x * blockDim.x + threadIdx.x;
  if (idx < NGRAPHS * EMB) buf[idx] *= inv[idx / EMB];
}

// fp32 [L][K][N] -> bf16 transposed+padded [L][N][Kpad]
__global__ void k_wconv(const float* __restrict__ src, __bf16* __restrict__ dst,
                        int K, int N, int Kpad, int tot) {
  int idx = blockIdx.x * blockDim.x + threadIdx.x;
  if (idx >= tot) return;
  int kp  = idx % Kpad;
  int rem = idx / Kpad;
  int n   = rem % N;
  int l   = rem / N;
  float v = (kp < K) ? src[((size_t)l * K + kp) * N + n] : 0.0f;
  dst[idx] = (__bf16)v;
}

// ---------------------------------------------------------------- atom encoder
struct AtomPtrs { const float* w[9]; };

__global__ __launch_bounds__(256)
void k_atom(const float* __restrict__ x, AtomPtrs aw, float* __restrict__ h) {
  const int AD[9]  = {119, 4, 12, 12, 10, 6, 6, 2, 2};
  const int OFF[9] = {0, 119, 123, 135, 147, 157, 163, 169, 171};
  const int TOT = 173;
  __shared__ float sp[8][176];
  __shared__ float sinv[8][9];
  int wid = threadIdx.x >> 5, lane = threadIdx.x & 31;
  int n = blockIdx.x * 8 + wid;
  bool act = (n < NNODES);

  for (int idx = lane; idx < TOT; idx += 32) {
    int i = 8;
#pragma unroll
    for (int j = 8; j >= 1; --j) if (idx < OFF[j]) i = j - 1;
    int k = idx - OFF[i];
    float e = 0.0f;
    if (act) {
      float col = x[(size_t)n * 9 + i];
      float tt  = col / (float)(AD[i] - 1);
      float mx  = fmaxf(col, 0.0f);
      e = __expf(tt * (float)k - mx);
    }
    sp[wid][idx] = e;
  }
  __syncthreads();
  if (lane < 9) {
    float s = 0.0f;
    for (int k = 0; k < AD[lane]; ++k) s += sp[wid][OFF[lane] + k];
    sinv[wid][lane] = (s > 0.0f) ? 1.0f / s : 0.0f;
  }
  __syncthreads();
  if (act) {
    for (int c = lane; c < EMB; c += 32) {
      float acc = 0.0f;
#pragma unroll
      for (int i = 0; i < 9; ++i) {
        const float* W = aw.w[i];
        float part = 0.0f;
        for (int k = 0; k < AD[i]; ++k)
          part += sp[wid][OFF[i] + k] * W[(size_t)k * EMB + c];
        acc += sinv[wid][i] * part;
      }
      h[(size_t)n * EMB + c] = acc;
    }
  }
}

// -------------------------------------------------- fused bond-encode + message
// Bond weights (13 x 300) staged to LDS once per block; warps are persistent
// and loop edges grid-stride.
__global__ __launch_bounds__(256)
void k_message(const float* __restrict__ h_in, const float* __restrict__ edge_attr,
               const int* __restrict__ srcI, const int* __restrict__ dstI,
               const float* __restrict__ bw0, const float* __restrict__ bw1,
               const float* __restrict__ bw2, float* __restrict__ agg) {
  __shared__ float sbw[13 * EMB];
  for (int i = threadIdx.x; i < 13 * EMB; i += 256) {
    int k = i / EMB, c = i - k * EMB;
    float v;
    if (k < 5)       v = bw0[(size_t)k * EMB + c];
    else if (k < 11) v = bw1[(size_t)(k - 5) * EMB + c];
    else             v = bw2[(size_t)(k - 11) * EMB + c];
    sbw[i] = v;
  }
  __syncthreads();

  int gwarp = (blockIdx.x * 256 + threadIdx.x) >> 5;
  int lane  = threadIdx.x & 31;
  int nwarp = gridDim.x * 8;

  for (int e = gwarp; e < NEDGES; e += nwarp) {
    int s = srcI[e], d = dstI[e];
    float p[13];
    {
      float col = edge_attr[(size_t)e * 3 + 0];
      float tt = col * 0.25f, mx = fmaxf(col, 0.0f), sum = 0.0f;
#pragma unroll
      for (int k = 0; k < 5; ++k) { p[k] = __expf(tt * k - mx); sum += p[k]; }
      float iv = 1.0f / sum;
#pragma unroll
      for (int k = 0; k < 5; ++k) p[k] *= iv;
    }
    {
      float col = edge_attr[(size_t)e * 3 + 1];
      float tt = col * 0.2f, mx = fmaxf(col, 0.0f), sum = 0.0f;
#pragma unroll
      for (int k = 0; k < 6; ++k) { p[5 + k] = __expf(tt * k - mx); sum += p[5 + k]; }
      float iv = 1.0f / sum;
#pragma unroll
      for (int k = 0; k < 6; ++k) p[5 + k] *= iv;
    }
    {
      float col = edge_attr[(size_t)e * 3 + 2];
      float mx = fmaxf(col, 0.0f), sum = 0.0f;
#pragma unroll
      for (int k = 0; k < 2; ++k) { p[11 + k] = __expf(col * k - mx); sum += p[11 + k]; }
      float iv = 1.0f / sum;
#pragma unroll
      for (int k = 0; k < 2; ++k) p[11 + k] *= iv;
    }

    const float* hs = h_in + (size_t)s * EMB;
    float* ag = agg + (size_t)d * EMB;
    for (int c = lane; c < EMB; c += 32) {
      float m = hs[c];
#pragma unroll
      for (int k = 0; k < 13; ++k) m += p[k] * sbw[k * EMB + c];
      atomicAdd(&ag[c], fmaxf(m, 0.0f));
    }
  }
}

// -------------------------------------------------------------- WMMA GEMM + BN
// out[M x N] = act( gam * ((A @ Bt^T + bias) * BN_SCALE) + bet )
//   A = alpha * rowscale[row] * Aa + Ab  (compile-time selected; alpha = 1+epsv[0])
//   Bt: bf16, [N][Kpad], zero-padded in K  (row n = column n of W)
// Double-buffered LDS; B tiles copied with async global->LDS DMA (ASYNCcnt);
// OOB rows/cols clamped (their C rows/cols are never stored).
template <bool HAS_AB, bool HAS_RS, bool HAS_EPS>
__global__ __launch_bounds__(256)
void k_gemm_bn(const float* __restrict__ Aa, const float* __restrict__ Ab,
               const float* __restrict__ rowscale, const float* __restrict__ epsv,
               const __bf16* __restrict__ Bt,
               const float* __restrict__ bias, const float* __restrict__ gam,
               const float* __restrict__ bet, float* __restrict__ out,
               int M, int K, int Kpad, int N, int relu) {
  __shared__ __bf16 sA[2][32][40];   // 2 x (32 rows x 32 K, +pad)
  __shared__ __bf16 sB[2][64][40];   // 2 x (64 cols x 32 K, +pad)
  const int t = threadIdx.x;
  const int lane = t & 31, wave = t >> 5;
  const int row0 = blockIdx.x * 32, col0 = blockIdx.y * 64;
  const int mt = (wave & 1) * 16;       // wave's M sub-tile
  const int nt = (wave >> 1) * 16;      // wave's N sub-tile
  const float alpha = HAS_EPS ? (1.0f + epsv[0]) : 1.0f;

  // fixed staging assignment (hoisted out of the K loop)
  const int ar  = t >> 3;               // A row in tile: 0..31
  const int ac4 = (t & 7) * 4;          // A col group: 0,4,...,28
  int agr = row0 + ar; if (agr > M - 1) agr = M - 1;   // clamp
  const float* aRow  = Aa + (size_t)agr * K;
  const float* abRow = HAS_AB ? (Ab + (size_t)agr * K) : nullptr;
  const float rs = alpha * (HAS_RS ? rowscale[agr] : 1.0f);

  const int bn_ = t >> 2;               // B row (W column) in tile: 0..63
  const int bk8 = (t & 3) * 8;          // K offset: 0,8,16,24
  int bgn = col0 + bn_; if (bgn > N - 1) bgn = N - 1;  // clamp
  const __bf16* bRow = Bt + (size_t)bgn * Kpad;

  v8f acc = {0.f, 0.f, 0.f, 0.f, 0.f, 0.f, 0.f, 0.f};

  // fragment geometry (fixed)
  const int fm  = mt + (lane & 15);
  const int fkb = (lane >> 4) * 8;
  const int fn  = nt + (lane & 15);
  const int fk2 = (lane >> 4) * 8;

  auto stageB = [&](int k0, int buf) {
    // async DMA: 16 bytes/lane, global -> LDS (tracked by ASYNCcnt)
    uint32_t lv = (uint32_t)(uintptr_t)(&sB[buf][bn_][bk8]);  // LDS offset in addr[31:0]
    uint64_t gv = (uint64_t)(uintptr_t)(bRow + k0 + bk8);
    asm volatile("global_load_async_to_lds_b128 %0, %1, off"
                 :: "v"(lv), "v"(gv) : "memory");
  };
  auto stageA = [&](int k0, int buf) {
    if (k0 + 32 <= K) {
      float4 av = *(const float4*)(aRow + k0 + ac4);      // 16B aligned (K%4==0)
      float x0 = av.x * rs, x1 = av.y * rs, x2 = av.z * rs, x3 = av.w * rs;
      if (HAS_AB) {
        float4 bv = *(const float4*)(abRow + k0 + ac4);
        x0 += bv.x; x1 += bv.y; x2 += bv.z; x3 += bv.w;
      }
      v4bf oa; oa[0] = (__bf16)x0; oa[1] = (__bf16)x1; oa[2] = (__bf16)x2; oa[3] = (__bf16)x3;
      *(v4bf*)(&sA[buf][ar][ac4]) = oa;
    } else {  // K tail: zero-fill beyond K (Bt already zero-padded)
#pragma unroll
      for (int j = 0; j < 4; ++j) {
        int gk = k0 + ac4 + j;
        float v = 0.0f;
        if (gk < K) {
          v = aRow[gk] * rs;
          if (HAS_AB) v += abRow[gk];
        }
        sA[buf][ar][ac4 + j] = (__bf16)v;
      }
    }
  };
  auto mma_step = [&](int buf) {
    v8bf a0 = *(const v8bf*)(&sA[buf][fm][fkb]);
    v8bf a1 = *(const v8bf*)(&sA[buf][fm][fkb + 16]);
    v16bf afr = __builtin_shufflevector(a0, a1, 0,1,2,3,4,5,6,7,8,9,10,11,12,13,14,15);
    v8bf b0 = *(const v8bf*)(&sB[buf][fn][fk2]);
    v8bf b1 = *(const v8bf*)(&sB[buf][fn][fk2 + 16]);
    v16bf bfr = __builtin_shufflevector(b0, b1, 0,1,2,3,4,5,6,7,8,9,10,11,12,13,14,15);
    acc = __builtin_amdgcn_wmma_f32_16x16x32_bf16(false, afr, false, bfr,
                                                  (short)0, acc, false, false);
  };

  const int ntiles = Kpad / 32;
  stageB(0, 0);
  stageA(0, 0);
  asm volatile("s_wait_asynccnt 0x0" ::: "memory");
  __syncthreads();
  for (int i = 0; i < ntiles; ++i) {
    const int cur = i & 1;
    if (i + 1 < ntiles) {         // stage next tile while computing current
      stageB((i + 1) * 32, cur ^ 1);
      stageA((i + 1) * 32, cur ^ 1);
    }
    mma_step(cur);
    asm volatile("s_wait_asynccnt 0x0" ::: "memory");
    __syncthreads();
  }

  // epilogue: lane holds column (lane&15), rows r + 8*(lane>=16)
  int col = col0 + nt + (lane & 15);
  int hi  = lane >> 4;
  if (col < N) {
    float g = gam[col], bb = bet[col], bi = bias[col];
#pragma unroll
    for (int r = 0; r < 8; ++r) {
      int row = row0 + mt + r + 8 * hi;
      if (row < M) {
        float v = g * ((acc[r] + bi) * BN_SCALE) + bb;
        if (relu) v = fmaxf(v, 0.0f);
        out[(size_t)row * N + col] = v;
      }
    }
  }
}

// ------------------------------------------------------------------- launcher
extern "C" void kernel_launch(void* const* d_in, const int* in_sizes, int n_in,
                              void* d_out, int out_size, void* d_ws, size_t ws_size,
                              hipStream_t stream) {
  (void)in_sizes; (void)n_in; (void)out_size; (void)ws_size;
  const float* x          = (const float*)d_in[0];
  const float* edge_attr  = (const float*)d_in[1];
  const int*   edge_index = (const int*)d_in[2];
  const int*   batch      = (const int*)d_in[3];
  AtomPtrs ap;
  for (int i = 0; i < 9; ++i) ap.w[i] = (const float*)d_in[4 + i];
  const float* bond_w0 = (const float*)d_in[13];
  const float* bond_w1 = (const float*)d_in[14];
  const float* bond_w2 = (const float*)d_in[15];
  const float* vn_emb  = (const float*)d_in[16];
  const float* eps     = (const float*)d_in[17];
  const float* W1   = (const float*)d_in[18];
  const float* b1   = (const float*)d_in[19];
  const float* bn1g = (const float*)d_in[20];
  const float* bn1b = (const float*)d_in[21];
  const float* W2   = (const float*)d_in[22];
  const float* b2   = (const float*)d_in[23];
  const float* bng  = (const float*)d_in[24];
  const float* bnb  = (const float*)d_in[25];
  const float* vW1  = (const float*)d_in[26];
  const float* vb1  = (const float*)d_in[27];
  const float* vbn1g= (const float*)d_in[28];
  const float* vbn1b= (const float*)d_in[29];
  const float* vW2  = (const float*)d_in[30];
  const float* vb2  = (const float*)d_in[31];
  const float* vbn2g= (const float*)d_in[32];
  const float* vbn2b= (const float*)d_in[33];

  // workspace layout
  char* ws = (char*)d_ws;
  size_t off = 0;
  auto alloc = [&](size_t bytes) -> void* {
    void* p = ws + off;
    off = (off + bytes + 255) & ~(size_t)255;
    return p;
  };
  float* h      = (float*)alloc((size_t)NNODES * EMB * 4);
  float* h_in   = (float*)alloc((size_t)NNODES * EMB * 4);
  float* agg    = (float*)alloc((size_t)NNODES * EMB * 4);
  float* z      = (float*)alloc((size_t)NNODES * HID * 4);
  float* vn     = (float*)alloc((size_t)NGRAPHS * EMB * 4);
  float* pooled = (float*)alloc((size_t)NGRAPHS * EMB * 4);
  float* vtmp   = (float*)alloc((size_t)NGRAPHS * HID * 4);
  float* counts = (float*)alloc((size_t)NGRAPHS * 4);
  float* invc   = (float*)alloc((size_t)NGRAPHS * 4);
  __bf16* W1bf  = (__bf16*)alloc((size_t)NLAYERS * HID * KPAD1 * 2);
  __bf16* W2bf  = (__bf16*)alloc((size_t)NLAYERS * EMB * KPAD2 * 2);
  __bf16* vW1bf = (__bf16*)alloc((size_t)(NLAYERS - 1) * HID * KPAD1 * 2);
  __bf16* vW2bf = (__bf16*)alloc((size_t)(NLAYERS - 1) * EMB * KPAD2 * 2);

  const int* srcI = edge_index;
  const int* dstI = edge_index + NEDGES;

  hipMemsetAsync(counts, 0, (size_t)NGRAPHS * 4, stream);
  k_count<<<(NNODES + 255) / 256, 256, 0, stream>>>(batch, counts);
  k_invcounts<<<(NGRAPHS + 255) / 256, 256, 0, stream>>>(counts, invc);
  {
    int tot1 = NLAYERS * HID * KPAD1;
    k_wconv<<<(tot1 + 255) / 256, 256, 0, stream>>>(W1, W1bf, EMB, HID, KPAD1, tot1);
    int tot2 = NLAYERS * EMB * KPAD2;
    k_wconv<<<(tot2 + 255) / 256, 256, 0, stream>>>(W2, W2bf, HID, EMB, KPAD2, tot2);
    int tot3 = (NLAYERS - 1) * HID * KPAD1;
    k_wconv<<<(tot3 + 255) / 256, 256, 0, stream>>>(vW1, vW1bf, EMB, HID, KPAD1, tot3);
    int tot4 = (NLAYERS - 1) * EMB * KPAD2;
    k_wconv<<<(tot4 + 255) / 256, 256, 0, stream>>>(vW2, vW2bf, HID, EMB, KPAD2, tot4);
  }

  k_atom<<<(NNODES + 7) / 8, 256, 0, stream>>>(x, ap, h);
  k_vn_init<<<(NGRAPHS * EMB + 255) / 256, 256, 0, stream>>>(vn_emb, vn);

  for (int l = 0; l < NLAYERS; ++l) {
    k_addvn<<<(NNODES * 75 + 255) / 256, 256, 0, stream>>>(h, vn, batch, h_in);
    hipMemsetAsync(agg, 0, (size_t)NNODES * EMB * 4, stream);
    k_message<<<1024, 256, 0, stream>>>(
        h_in, edge_attr, srcI, dstI,
        bond_w0 + (size_t)l * 5 * EMB, bond_w1 + (size_t)l * 6 * EMB,
        bond_w2 + (size_t)l * 2 * EMB, agg);

    // z = relu(bn1(((1+eps)*h_in + agg) @ W1 + b1))
    {
      dim3 grid(NNODES / 32, (HID + 63) / 64);
      k_gemm_bn<true, false, true><<<grid, 256, 0, stream>>>(
          h_in, agg, nullptr, eps + l, W1bf + (size_t)l * HID * KPAD1,
          b1 + (size_t)l * HID, bn1g + (size_t)l * HID, bn1b + (size_t)l * HID,
          z, NNODES, EMB, KPAD1, HID, 1);
    }
    // h = bn(z @ W2 + b2), relu on all but last layer
    {
      dim3 grid(NNODES / 32, (EMB + 63) / 64);
      k_gemm_bn<false, false, false><<<grid, 256, 0, stream>>>(
          z, nullptr, nullptr, nullptr, W2bf + (size_t)l * EMB * KPAD2,
          b2 + (size_t)l * EMB, bng + (size_t)l * EMB, bnb + (size_t)l * EMB,
          h, NNODES, HID, KPAD2, EMB, (l < NLAYERS - 1) ? 1 : 0);
    }
    if (l < NLAYERS - 1) {
      hipMemsetAsync(pooled, 0, (size_t)NGRAPHS * EMB * 4, stream);
      k_pool<<<(NNODES * 75 + 255) / 256, 256, 0, stream>>>(h_in, batch, pooled);
      // vtmp = relu(vbn1((pooled/counts + vn) @ vW1 + vb1))
      dim3 g1((NGRAPHS + 31) / 32, (HID + 63) / 64);
      k_gemm_bn<true, true, false><<<g1, 256, 0, stream>>>(
          pooled, vn, invc, nullptr, vW1bf + (size_t)l * HID * KPAD1,
          vb1 + (size_t)l * HID, vbn1g + (size_t)l * HID, vbn1b + (size_t)l * HID,
          vtmp, NGRAPHS, EMB, KPAD1, HID, 1);
      // vn = relu(vbn2(vtmp @ vW2 + vb2))
      dim3 g2((NGRAPHS + 31) / 32, (EMB + 63) / 64);
      k_gemm_bn<false, false, false><<<g2, 256, 0, stream>>>(
          vtmp, nullptr, nullptr, nullptr, vW2bf + (size_t)l * EMB * KPAD2,
          vb2 + (size_t)l * EMB, vbn2g + (size_t)l * EMB, vbn2b + (size_t)l * EMB,
          vn, NGRAPHS, HID, KPAD2, EMB, 1);
    }
  }

  // final graph mean-pool of h -> d_out
  hipMemsetAsync(d_out, 0, (size_t)NGRAPHS * EMB * 4, stream);
  k_pool<<<(NNODES * 75 + 255) / 256, 256, 0, stream>>>(h, batch, (float*)d_out);
  k_scale_rows<<<(NGRAPHS * EMB + 255) / 256, 256, 0, stream>>>((float*)d_out, invc);
}
